// BSQAttention_68899865362464
// MI455X (gfx1250) — matrix operand order repaired and checked by reading, not compiled, not run on hardware
//
#include <hip/hip_runtime.h>
#include <hip/hip_bf16.h>

typedef __attribute__((ext_vector_type(16))) _Float16 v16h;
typedef __attribute__((ext_vector_type(8)))  _Float16 v8h;
typedef __attribute__((ext_vector_type(8)))  float    v8f;

#define CODE_SIZE 8
#define KC 256          // 2^CODE_SIZE buckets
#define NHEAD 16
#define HD 64           // head dim
#define CDIM 1024       // channels

// ---------------------------------------------------------------------------
// K0: zero the scatter accumulators (cod_v + counts live contiguously in ws)
// ---------------------------------------------------------------------------
__global__ void k_zero(float* __restrict__ p, int n) {
    int i = blockIdx.x * blockDim.x + threadIdx.x;
    if (i < n) p[i] = 0.f;
}

// ---------------------------------------------------------------------------
// K1: per-kv-token code -> bucket index, then scatter-add v and counts.
// One wave (32 lanes) per kv row. 8 dot products of length 1024, butterfly
// reduce, sign-pack to 8-bit index, then 32 f32 atomics per lane for v.
// ---------------------------------------------------------------------------
__global__ void k_code_scatter(const float* __restrict__ kmat,
                               const float* __restrict__ vmat,
                               const float* __restrict__ Wc,
                               const float* __restrict__ bc,
                               float* __restrict__ cod_v,
                               float* __restrict__ counts,
                               int Lk, int nrows) {
    const int wave = threadIdx.x >> 5;
    const int lane = threadIdx.x & 31;
    const int row = blockIdx.x * (blockDim.x >> 5) + wave;
    if (row >= nrows) return;
    const int b = row / Lk;

    const float* kr = kmat + (size_t)row * CDIM;
    float acc[CODE_SIZE];
#pragma unroll
    for (int j = 0; j < CODE_SIZE; j++) acc[j] = 0.f;

    for (int i = 0; i < CDIM / 32; i++) {
        int c = i * 32 + lane;            // coalesced
        float kv = kr[c];
#pragma unroll
        for (int j = 0; j < CODE_SIZE; j++) acc[j] += kv * Wc[j * CDIM + c];
    }
#pragma unroll
    for (int j = 0; j < CODE_SIZE; j++) {
#pragma unroll
        for (int d = 1; d < 32; d <<= 1) acc[j] += __shfl_xor(acc[j], d, 32);
    }
    // sign bits, MSB-first (code_base = 2^(7..0)); normalization keeps signs
    int idx = 0;
#pragma unroll
    for (int j = 0; j < CODE_SIZE; j++)
        if (acc[j] + bc[j] >= 0.f) idx |= (1 << (CODE_SIZE - 1 - j));

    float* dst = cod_v + ((size_t)b * KC + idx) * CDIM;
    const float* vr = vmat + (size_t)row * CDIM;
    for (int i = 0; i < CDIM / 32; i++) {
        int c = i * 32 + lane;
        atomicAdd(dst + c, vr[c]);        // global_atomic_add_f32
    }
    if (lane == 0) atomicAdd(counts + b * KC + idx, 1.0f);
}

// ---------------------------------------------------------------------------
// K2: dense codebook keys for all 256 codes, stored f16 row-major [bucket][C]
// cod_k[n][c] = sum_j (bit_j(n) ? codebook[j][c] : codebook[8+j][c])
// ---------------------------------------------------------------------------
__global__ void k_codk(const float* __restrict__ codebook,
                       _Float16* __restrict__ codk) {
    int t = blockIdx.x * blockDim.x + threadIdx.x;   // exactly 256*1024 threads
    int n = t >> 10;
    int c = t & (CDIM - 1);
    float s = 0.f;
#pragma unroll
    for (int j = 0; j < CODE_SIZE; j++) {
        int bit = (n >> (CODE_SIZE - 1 - j)) & 1;
        s += bit ? codebook[j * CDIM + c] : codebook[(CODE_SIZE + j) * CDIM + c];
    }
    codk[t] = (_Float16)s;
}

// ---------------------------------------------------------------------------
// K3: transpose cod_v f32 [b][bucket][C] -> f16 vT [b][c][bucket]
// (so GEMM2 B-fragments are contiguous along the bucket/K dimension)
// ---------------------------------------------------------------------------
__global__ void k_vT(const float* __restrict__ cod_v,
                     _Float16* __restrict__ vT) {
    int t = blockIdx.x * blockDim.x + threadIdx.x;   // B * 1024 * 256 threads
    int n = t & (KC - 1);
    int c = (t >> 8) & (CDIM - 1);
    int b = t >> 18;
    vT[t] = (_Float16)cod_v[((size_t)b * KC + n) * CDIM + c];
}

// ---------------------------------------------------------------------------
// K4: bucketed attention. One wave per (b, head, 16-query block).
// GEMM1: S[16,256] = Q[16,64] @ Khᵀ via 32 x v_wmma_f32_16x16x32_f16
// softmax-less ratio: e = gate*exp(s-max); out = (e@Vh)/(e@cnt)
// GEMM2: num[16,64] = E[16,256] @ Vh via 32 more WMMAs (E via LDS re-layout)
// ---------------------------------------------------------------------------
__launch_bounds__(128)
__global__ void k_attn(const float* __restrict__ q,
                       const _Float16* __restrict__ codk,
                       const _Float16* __restrict__ vT,
                       const float* __restrict__ counts,
                       float* __restrict__ out,
                       int Lq, int ntasks) {
    __shared__ _Float16 smem[4 * 16 * KC];   // 8KB per wave, 4 waves
    const int lane = threadIdx.x & 31;
    const int wave = threadIdx.x >> 5;
    int task = blockIdx.x * 4 + wave;
    if (task >= ntasks) task = ntasks - 1;   // clamp: keeps barrier convergent

    const int qblocks = Lq >> 4;
    const int b  = task / (NHEAD * qblocks);
    const int r0 = task % (NHEAD * qblocks);
    const int h  = r0 / qblocks;
    const int qb = r0 % qblocks;

    const int m  = lane & 15;   // row (A) / column (B,D) index
    const int hs = lane >> 4;   // lane-half select

    // ---- A fragments from Q (f32 -> f16), K-chunks kc=0 and kc=32 --------
    // 16-bit A 16x32 layout: lane half 0 -> K {kc..kc+7, kc+16..kc+23},
    //                        lane half 1 -> K {kc+8..kc+15, kc+24..kc+31}
    const float* qrow = q + ((size_t)b * Lq + qb * 16 + m) * CDIM + h * HD;
    v16h aq[2];
#pragma unroll
    for (int kk = 0; kk < 2; kk++) {
        const float* p0 = qrow + kk * 32 + hs * 8;
        const float* p1 = p0 + 16;
        union { v16h v; _Float16 e[16]; } u;
#pragma unroll
        for (int e = 0; e < 8; e++) u.e[e]     = (_Float16)p0[e];
#pragma unroll
        for (int e = 0; e < 8; e++) u.e[8 + e] = (_Float16)p1[e];
        aq[kk] = u.v;
    }

    // ---- GEMM1: scores over 16 bucket tiles ------------------------------
    v8f S[16];
#pragma unroll
    for (int nt = 0; nt < 16; nt++) {
        v8f c = {0.f, 0.f, 0.f, 0.f, 0.f, 0.f, 0.f, 0.f};
#pragma unroll
        for (int kk = 0; kk < 2; kk++) {
            // B 32x16 layout: lane = column n (bucket), K {8hs..8hs+7, 16+8hs..16+8hs+7}
            const _Float16* kp = codk + ((size_t)(nt * 16 + m)) * CDIM
                               + h * HD + kk * 32 + hs * 8;
            union { v16h v; v8h half2[2]; } ub;
            ub.half2[0] = *(const v8h*)kp;
            ub.half2[1] = *(const v8h*)(kp + 16);
            v16h bk = ub.v;
            c = __builtin_amdgcn_wmma_f32_16x16x32_f16(
                    false, aq[kk], false, bk, (short)0, c, false, false);
        }
        S[nt] = c * 0.125f;   // scale = d^-0.5 = 1/8
    }

    // ---- bucket counts for this lane's column across all tiles -----------
    float cnt[16];
#pragma unroll
    for (int t = 0; t < 16; t++) cnt[t] = counts[b * KC + t * 16 + m];

    // ---- row max (butterfly within each 16-lane half) --------------------
    float mx[8], den[8];
#pragma unroll
    for (int r = 0; r < 8; r++) {
        float loc = S[0][r];
#pragma unroll
        for (int t = 1; t < 16; t++) loc = fmaxf(loc, S[t][r]);
#pragma unroll
        for (int d = 1; d < 16; d <<= 1) loc = fmaxf(loc, __shfl_xor(loc, d, 32));
        mx[r] = loc;
        den[r] = 0.f;
    }

    // ---- e = gate * exp(s - max);  den = e @ counts ----------------------
#pragma unroll
    for (int t = 0; t < 16; t++) {
        float gate = (cnt[t] > 0.f) ? 1.f : 0.f;
#pragma unroll
        for (int r = 0; r < 8; r++) {
            float e = gate * __expf(S[t][r] - mx[r]);
            S[t][r] = e;
            den[r] += e * cnt[t];
        }
    }
#pragma unroll
    for (int r = 0; r < 8; r++) {
#pragma unroll
        for (int d = 1; d < 16; d <<= 1) den[r] += __shfl_xor(den[r], d, 32);
    }

    // ---- re-layout E through LDS into A-fragment form --------------------
    _Float16* ep = smem + wave * 16 * KC;  // [16 rows][256 buckets] f16
#pragma unroll
    for (int t = 0; t < 16; t++) {
#pragma unroll
        for (int r = 0; r < 8; r++)
            ep[(r + 8 * hs) * KC + t * 16 + m] = (_Float16)S[t][r];
    }
    __syncthreads();

    // ---- GEMM2: num[16,64] = E[16,256] @ Vh[256,64] ----------------------
    v8f zero8 = {0.f, 0.f, 0.f, 0.f, 0.f, 0.f, 0.f, 0.f};
    v8f N[4] = {zero8, zero8, zero8, zero8};
    const _Float16* vbase = vT + ((size_t)(b * NHEAD + h) * HD) * KC;
#pragma unroll
    for (int kc8 = 0; kc8 < 8; kc8++) {     // K = 32 buckets per step
        union { v16h v; v8h half2[2]; } a;
        const _Float16* ap = ep + m * KC + kc8 * 32 + hs * 8;
        a.half2[0] = *(const v8h*)ap;
        a.half2[1] = *(const v8h*)(ap + 16);
#pragma unroll
        for (int dt = 0; dt < 4; dt++) {    // 4 tiles of 16 along head dim
            const _Float16* vp = vbase + ((size_t)(dt * 16 + m)) * KC
                               + kc8 * 32 + hs * 8;
            union { v16h v; v8h half2[2]; } ub;
            ub.half2[0] = *(const v8h*)vp;
            ub.half2[1] = *(const v8h*)(vp + 16);
            v16h bv = ub.v;
            N[dt] = __builtin_amdgcn_wmma_f32_16x16x32_f16(
                        false, a.v, false, bv, (short)0, N[dt], false, false);
        }
    }

    // ---- out = num / den -------------------------------------------------
    float* obase = out + ((size_t)b * Lq + qb * 16) * CDIM + h * HD;
#pragma unroll
    for (int dt = 0; dt < 4; dt++) {
#pragma unroll
        for (int r = 0; r < 8; r++) {
            obase[(size_t)(r + 8 * hs) * CDIM + dt * 16 + m] = N[dt][r] / den[r];
        }
    }
}

// ---------------------------------------------------------------------------
extern "C" void kernel_launch(void* const* d_in, const int* in_sizes, int n_in,
                              void* d_out, int out_size, void* d_ws, size_t ws_size,
                              hipStream_t stream) {
    const float* q        = (const float*)d_in[0];
    const float* k        = (const float*)d_in[1];
    const float* v        = (const float*)d_in[2];
    const float* Wc       = (const float*)d_in[3];
    const float* bc       = (const float*)d_in[4];
    const float* codebook = (const float*)d_in[5];
    // d_in[6]=lengths, d_in[7]=inv_lengths: equal per sample; shapes via in_sizes
    const int B  = in_sizes[6];
    const int Lq = in_sizes[0] / (B * CDIM);
    const int Lk = in_sizes[1] / (B * CDIM);
    float* out = (float*)d_out;

    // workspace layout (all offsets 32B aligned)
    float*    cod_v  = (float*)d_ws;                          // B*256*1024 f32
    float*    counts = cod_v + (size_t)B * KC * CDIM;         // B*256 f32
    _Float16* codk   = (_Float16*)(counts + (size_t)B * KC);  // 256*1024 f16
    _Float16* vT     = codk + (size_t)KC * CDIM;              // B*16*64*256 f16
    (void)ws_size; (void)n_in; (void)out_size;

    int nz = B * KC * CDIM + B * KC;
    k_zero<<<(nz + 255) / 256, 256, 0, stream>>>(cod_v, nz);

    int nrows = B * Lk;
    k_code_scatter<<<(nrows + 7) / 8, 256, 0, stream>>>(k, v, Wc, bc,
                                                        cod_v, counts, Lk, nrows);

    k_codk<<<(KC * CDIM) / 256, 256, 0, stream>>>(codebook, codk);

    k_vT<<<(B * CDIM * KC) / 256, 256, 0, stream>>>(cod_v, vT);

    int ntasks = B * NHEAD * (Lq / 16);
    k_attn<<<(ntasks + 3) / 4, 128, 0, stream>>>(q, codk, vT, counts, out,
                                                 Lq, ntasks);
}
